// SemanticAwareTransitions_9826885173968
// MI455X (gfx1250) — matrix operand order repaired and checked by reading, not compiled, not run on hardware
//
#include <hip/hip_runtime.h>
#include <hip/hip_bf16.h>

typedef __attribute__((ext_vector_type(16))) _Float16 v16h;
typedef __attribute__((ext_vector_type(8)))  float    v8f;
typedef __attribute__((ext_vector_type(4)))  float    v4f;

// Problem constants
constexpr int cV = 512, cD = 256, cH = 8, cDH = 32, cFF = 2048, cL = 2;
constexpr int cB = 4, cS = 128, cBS = cB * cS;   // 512 tokens
#define LN_EPS 1e-5f

// ---------------------------------------------------------------------------
// CDNA5 async global->LDS copy (ASYNCcnt-tracked, no VGPR round trip).
// GV addressing mode: 64-bit vaddr in VGPR pair, saddr = off.
// ---------------------------------------------------------------------------
__device__ __forceinline__ void async_copy_b128(void* lds_ptr, const void* gptr)
{
    unsigned lds_off = (unsigned)(unsigned long long)lds_ptr;  // LDS byte addr = low 32 bits
    asm volatile("global_load_async_to_lds_b128 %0, %1, off"
                 :: "v"(lds_off), "v"(gptr) : "memory");
}
__device__ __forceinline__ void wait_async0()
{
    asm volatile("s_wait_asynccnt 0" ::: "memory");
}

// ---------------------------------------------------------------------------
// Generic batched WMMA GEMM:  C = act( alpha * A@B + bias )
//   A: [M,K] lda, per-batch base = zo*aO + zi*aI  (z=blockIdx.z, zo=z/inner, zi=z%inner)
//   B: [K,N] ldb; TRANSB: B(k,n) = b[n*ldb + k]
// Block: 256 threads = 8 waves; block tile 64(M) x 32(N); wave tile 16x16.
// K-loop is double-buffered through LDS with async-to-LDS prefetch.
// Requires M%64==0, N%32==0, K%32==0, lda/ldb%4==0 (true for every call here).
// ---------------------------------------------------------------------------
template <int TRANSB>
__global__ __launch_bounds__(256)
void wmma_gemm(const float* __restrict__ A, int lda, long aO, long aI,
               const float* __restrict__ B, int ldb, long bO, long bI,
               float* __restrict__ C, int ldc, long cO, long cI,
               const float* __restrict__ bias,
               int N, int K, int innerCount, float alpha, int relu)
{
    // Row stride 36 floats: 144B keeps b128 stores aligned; 36 vs 64 banks gives
    // 16 distinct banks for the 16-lane strided fragment reads.
    __shared__ float As[2][64][36];
    __shared__ float Bs[2][32][36];   // [k][n] normally, [n][k] when TRANSB

    const int z  = blockIdx.z;
    const int zo = z / innerCount, zi = z % innerCount;
    const float* a = A + (long)zo * aO + (long)zi * aI;
    const float* b = B + (long)zo * bO + (long)zi * bI;
    float*       c = C + (long)zo * cO + (long)zi * cI;

    const int bM   = blockIdx.y * 64;
    const int bN   = blockIdx.x * 32;
    const int tid  = threadIdx.x;
    const int lane = tid & 31;
    const int w    = tid >> 5;
    const int half = lane >> 4;      // 0: lanes 0-15, 1: lanes 16-31
    const int l15  = lane & 15;
    const int tm   = (w & 3) * 16;   // wave's M offset within block tile
    const int tn   = (w >> 2) * 16;  // wave's N offset within block tile

    // Per-thread staging slots (128-bit granules)
    const int ar = tid >> 3, ac = (tid & 7) * 4;   // A: rows ar and ar+32, cols ac..ac+3
    const int br = tid >> 3, bc = (tid & 7) * 4;   // B: one 4-wide granule

    auto issue_tile = [&](int buf, int k0) {
        async_copy_b128(&As[buf][ar][ac],      a + (long)(bM + ar)      * lda + k0 + ac);
        async_copy_b128(&As[buf][ar + 32][ac], a + (long)(bM + ar + 32) * lda + k0 + ac);
        if (TRANSB)   // Bs[n][k] <- contiguous along k in global
            async_copy_b128(&Bs[buf][br][bc], b + (long)(bN + br) * ldb + k0 + bc);
        else          // Bs[k][n] <- contiguous along n in global
            async_copy_b128(&Bs[buf][br][bc], b + (long)(k0 + br) * ldb + bN + bc);
    };

    v8f acc = {};
    int cur = 0;
    issue_tile(0, 0);

    for (int k0 = 0; k0 < K; k0 += 32) {
        wait_async0();        // my async writes to As/Bs[cur] have landed
        __syncthreads();      // everyone's have landed; prev buffer fully consumed

        if (k0 + 32 < K) issue_tile(cur ^ 1, k0 + 32);   // prefetch next K-tile

        // Pack A fragment: contiguous k order, same as B:
        //   lanes 0-15 -> K 0..15; lanes 16-31 -> K 16..31
        v16h af, bf;
        const int m = tm + l15;
        #pragma unroll
        for (int v = 0; v < 8; ++v) {
            int ka = 2 * v + 16 * half;
            af[2 * v]     = (_Float16)As[cur][m][ka];
            af[2 * v + 1] = (_Float16)As[cur][m][ka + 1];
        }
        // Pack B fragment: lanes 0-15 hold K=0..15, lanes 16-31 hold K=16..31
        const int n = tn + l15;
        #pragma unroll
        for (int e = 0; e < 16; ++e)
            bf[e] = (_Float16)(TRANSB ? Bs[cur][n][16 * half + e]
                                      : Bs[cur][16 * half + e][n]);

        acc = __builtin_amdgcn_wmma_f32_16x16x32_f16(
                  false, af, false, bf, (short)0, acc, false, false);
        cur ^= 1;
    }

    // C/D layout: VGPR v -> row v + 8*half; lane&15 -> col.
    // bias[col] is loop-invariant across the 8 rows: load it exactly once.
    const int col = bN + tn + l15;
    const float badd = bias ? bias[col] : 0.0f;
    #pragma unroll
    for (int v = 0; v < 8; ++v) {
        int row = bM + tm + v + 8 * half;
        float val = acc[v] * alpha + badd;
        if (relu) val = fmaxf(val, 0.0f);
        c[(long)row * ldc + col] = val;
    }
}

// ---------------------------------------------------------------------------
// Embedding gather: X[t, :] = emb[x_t[t], :]
// ---------------------------------------------------------------------------
__global__ __launch_bounds__(256)
void embed_kernel(const int* __restrict__ xt, const float* __restrict__ emb,
                  float* __restrict__ X)
{
    int t = blockIdx.x, d = threadIdx.x;
    X[t * cD + d] = emb[(long)xt[t] * cD + d];
}

// ---------------------------------------------------------------------------
// out = LayerNorm(x + y) * s + b    (biased variance, matches jnp.var)
// One wave per 256-wide row; 8 rows per 256-thread block.
// ---------------------------------------------------------------------------
__global__ __launch_bounds__(256)
void ln_residual_kernel(const float* __restrict__ x, const float* __restrict__ y,
                        const float* __restrict__ s, const float* __restrict__ bb,
                        float* __restrict__ out)
{
    int row  = blockIdx.x * 8 + (threadIdx.x >> 5);
    int lane = threadIdx.x & 31;
    float vals[8], sum = 0.f, sq = 0.f;
    #pragma unroll
    for (int i = 0; i < 8; ++i) {
        int cidx = lane + i * 32;
        float v = x[(long)row * cD + cidx] + y[(long)row * cD + cidx];
        vals[i] = v; sum += v; sq += v * v;
    }
    for (int o = 16; o > 0; o >>= 1) {
        sum += __shfl_xor(sum, o);
        sq  += __shfl_xor(sq, o);
    }
    float mean = sum * (1.0f / cD);
    float var  = sq * (1.0f / cD) - mean * mean;
    float inv  = rsqrtf(var + LN_EPS);
    #pragma unroll
    for (int i = 0; i < 8; ++i) {
        int cidx = lane + i * 32;
        out[(long)row * cD + cidx] = (vals[i] - mean) * inv * s[cidx] + bb[cidx];
    }
}

// ---------------------------------------------------------------------------
// Row softmax (optionally scaled by 1/temperature before exp).
// One wave per row; cols must be multiple of 32, <= 512.
// ---------------------------------------------------------------------------
__global__ __launch_bounds__(256)
void softmax_rows_kernel(const float* __restrict__ in, float* __restrict__ out,
                         int cols, const float* __restrict__ tptr)
{
    int row  = blockIdx.x * 8 + (threadIdx.x >> 5);
    int lane = threadIdx.x & 31;
    float invT = tptr ? (1.0f / tptr[0]) : 1.0f;
    float buf[16];
    int nIt = cols >> 5;
    float mx = -1e30f;
    for (int i = 0; i < nIt; ++i) {
        float v = in[(long)row * cols + lane + i * 32] * invT;
        buf[i] = v; mx = fmaxf(mx, v);
    }
    for (int o = 16; o > 0; o >>= 1) mx = fmaxf(mx, __shfl_xor(mx, o));
    float sum = 0.f;
    for (int i = 0; i < nIt; ++i) { buf[i] = expf(buf[i] - mx); sum += buf[i]; }
    for (int o = 16; o > 0; o >>= 1) sum += __shfl_xor(sum, o);
    float r = 1.0f / sum;
    for (int i = 0; i < nIt; ++i)
        out[(long)row * cols + lane + i * 32] = buf[i] * r;
}

// ---------------------------------------------------------------------------
// out[b,s,i,:] = tp[b,s,:] for all i.  512 MB of NT stores (bypass L2: output
// is 512MB >> 192MB L2, while the resident working set stays cached).
// With a stride-256 loop over a 128-entry v4f row, f % 128 == tid % 128 is
// constant per thread: each thread loads its 16B value once and then issues
// pure back-to-back NT b128 stores -- no LDS traffic in the streaming loop.
// ---------------------------------------------------------------------------
__global__ __launch_bounds__(256)
void broadcast_kernel(const float* __restrict__ tp, float* __restrict__ out)
{
    const int bs = blockIdx.x;
    const v4f* src = (const v4f*)(tp + (long)bs * cV);
    const v4f  val = src[threadIdx.x & (cV / 4 - 1)];   // this thread's column group
    v4f* dst = (v4f*)(out + (long)bs * cV * cV);
    const int total = cV * cV / 4;                       // 65536 float4 per (b,s)
    for (int f = threadIdx.x; f < total; f += 256)
        __builtin_nontemporal_store(val, dst + f);
}

// ---------------------------------------------------------------------------
extern "C" void kernel_launch(void* const* d_in, const int* in_sizes, int n_in,
                              void* d_out, int out_size, void* d_ws, size_t ws_size,
                              hipStream_t stream)
{
    (void)in_sizes; (void)n_in; (void)out_size; (void)ws_size;

    const int*   x_t  = (const int*)  d_in[0];
    const float* emb  = (const float*)d_in[1];
    const float* temp = (const float*)d_in[2];
    const float* Wq   = (const float*)d_in[3];
    const float* Wk   = (const float*)d_in[4];
    const float* Wv   = (const float*)d_in[5];
    const float* Wo   = (const float*)d_in[6];
    const float* bq   = (const float*)d_in[7];
    const float* bk   = (const float*)d_in[8];
    const float* bv   = (const float*)d_in[9];
    const float* bo   = (const float*)d_in[10];
    const float* W1   = (const float*)d_in[11];
    const float* b1   = (const float*)d_in[12];
    const float* W2   = (const float*)d_in[13];
    const float* b2   = (const float*)d_in[14];
    const float* ln1s = (const float*)d_in[15];
    const float* ln1b = (const float*)d_in[16];
    const float* ln2s = (const float*)d_in[17];
    const float* ln2b = (const float*)d_in[18];
    float* out = (float*)d_out;

    // Workspace layout (floats); total ~12.1 MB
    float* ws  = (float*)d_ws;
    float* X0  = ws;                 // 512*256
    float* X1  = X0 + cBS * cD;
    float* Qb  = X1 + cBS * cD;
    float* Kb  = Qb + cBS * cD;
    float* Vb  = Kb + cBS * cD;
    float* Ob  = Vb + cBS * cD;
    float* Tmp = Ob + cBS * cD;
    float* Sc  = Tmp + cBS * cD;     // 32*128*128
    float* Fb  = Sc + cB * cH * cS * cS;   // 512*2048
    float* Lg  = Fb + cBS * cFF;     // 512*512
    float* Tp  = Lg + cBS * cV;      // 512*512

    const dim3 blk(256);
    const long sBD = (long)cS * cD;        // per-b stride in Q/K/V/O buffers
    const long sHS = (long)cH * cS * cS;   // per-b stride in scores
    const long sSS = (long)cS * cS;        // per-h stride in scores
    const float iscale = 0.17677669529663687f;   // 1/sqrt(32)

    embed_kernel<<<cBS, blk, 0, stream>>>(x_t, emb, X0);

    float* Xc = X0;   // current activations
    float* Xn = X1;   // post-attention activations

    for (int l = 0; l < cL; ++l) {
        const float* wq = Wq + (long)l * cD * cD;
        const float* wk = Wk + (long)l * cD * cD;
        const float* wv = Wv + (long)l * cD * cD;
        const float* wo = Wo + (long)l * cD * cD;
        const float* w1 = W1 + (long)l * cD * cFF;
        const float* w2 = W2 + (long)l * cFF * cD;

        // Q/K/V projections: [512,256] @ [256,256] + bias
        wmma_gemm<0><<<dim3(8, 8, 1), blk, 0, stream>>>(
            Xc, cD, 0L, 0L, wq, cD, 0L, 0L, Qb, cD, 0L, 0L,
            bq + l * cD, cD, cD, 1, 1.0f, 0);
        wmma_gemm<0><<<dim3(8, 8, 1), blk, 0, stream>>>(
            Xc, cD, 0L, 0L, wk, cD, 0L, 0L, Kb, cD, 0L, 0L,
            bk + l * cD, cD, cD, 1, 1.0f, 0);
        wmma_gemm<0><<<dim3(8, 8, 1), blk, 0, stream>>>(
            Xc, cD, 0L, 0L, wv, cD, 0L, 0L, Vb, cD, 0L, 0L,
            bv + l * cD, cD, cD, 1, 1.0f, 0);

        // scores[b,h] = Q[b,:,h] @ K[b,:,h]^T * 1/sqrt(DH)   (32 batches)
        wmma_gemm<1><<<dim3(4, 2, cB * cH), blk, 0, stream>>>(
            Qb, cD, sBD, (long)cDH, Kb, cD, sBD, (long)cDH,
            Sc, cS, sHS, sSS, nullptr, cS, cDH, cH, iscale, 0);

        // softmax over keys: 4096 rows of 128
        softmax_rows_kernel<<<cB * cH * cS / 8, blk, 0, stream>>>(Sc, Sc, cS, nullptr);

        // O[b,:,h] = attn[b,h] @ V[b,:,h]
        wmma_gemm<0><<<dim3(1, 2, cB * cH), blk, 0, stream>>>(
            Sc, cS, sHS, sSS, Vb, cD, sBD, (long)cDH,
            Ob, cD, sBD, (long)cDH, nullptr, cDH, cS, cH, 1.0f, 0);

        // Output projection, then post-norm with residual
        wmma_gemm<0><<<dim3(8, 8, 1), blk, 0, stream>>>(
            Ob, cD, 0L, 0L, wo, cD, 0L, 0L, Tmp, cD, 0L, 0L,
            bo + l * cD, cD, cD, 1, 1.0f, 0);
        ln_residual_kernel<<<cBS / 8, blk, 0, stream>>>(
            Xc, Tmp, ln1s + l * cD, ln1b + l * cD, Xn);

        // FFN: relu(Xn @ W1 + b1) @ W2 + b2, then post-norm with residual
        wmma_gemm<0><<<dim3(64, 8, 1), blk, 0, stream>>>(
            Xn, cD, 0L, 0L, w1, cFF, 0L, 0L, Fb, cFF, 0L, 0L,
            b1 + l * cFF, cFF, cD, 1, 1.0f, 1);
        wmma_gemm<0><<<dim3(8, 8, 1), blk, 0, stream>>>(
            Fb, cFF, 0L, 0L, w2, cD, 0L, 0L, Tmp, cD, 0L, 0L,
            b2 + l * cD, cD, cFF, 1, 1.0f, 0);
        ln_residual_kernel<<<cBS / 8, blk, 0, stream>>>(
            Xn, Tmp, ln2s + l * cD, ln2b + l * cD, Xc);
    }

    // logits = X @ emb^T   (transB: emb is [V,D])
    wmma_gemm<1><<<dim3(16, 8, 1), blk, 0, stream>>>(
        Xc, cD, 0L, 0L, emb, cD, 0L, 0L, Lg, cV, 0L, 0L,
        nullptr, cV, cD, 1, 1.0f, 0);

    // tp = softmax(logits / T), rows 512 x cols 512
    softmax_rows_kernel<<<cBS / 8, blk, 0, stream>>>(Lg, Tp, cV, temp);

    // out[b,s,i,j] = tp[b,s,j]  -- 512 MB of NT streaming stores
    broadcast_kernel<<<cBS, blk, 0, stream>>>(Tp, out);
}
